// TwoHeadNeighborhoodAttentionAggregator_6923487281747
// MI455X (gfx1250) — matrix operand order repaired and checked by reading, not compiled, not run on hardware
//
#include <hip/hip_runtime.h>
#include <hip/hip_bf16.h>

// ---------------------------------------------------------------------------
// TwoHeadNeighborhoodAttentionAggregator for MI455X (gfx1250, wave32, WMMA)
// B=16384, N=200000, K=16, E=256, H=2, A=64, HD=128
// ---------------------------------------------------------------------------

typedef __attribute__((ext_vector_type(16))) __bf16 v16bf;
typedef __attribute__((ext_vector_type(8)))  __bf16 v8bf;
typedef __attribute__((ext_vector_type(8)))  float  v8f;

#define TB 16          // batch rows per block
#define NTHREADS 256   // 8 wave32

__device__ __forceinline__ v16bf cat8(v8bf lo, v8bf hi) {
  return __builtin_shufflevector(lo, hi, 0,1,2,3,4,5,6,7,8,9,10,11,12,13,14,15);
}

// A-fragment (16x32 bf16) from a contiguous row-major LDS tile.
// m = lane&15 ; k already includes half*8 lane offset. Chunks at k and k+16.
__device__ __forceinline__ v16bf loadA_contig(const __bf16* base, int ld, int m, int k) {
  const __bf16* p = base + m * ld + k;
  v8bf lo = *(const v8bf*)p;
  v8bf hi = *(const v8bf*)(p + 16);
  return cat8(lo, hi);
}

// A-fragment for pair = [center(256) | neighbor(256)] split rows.
__device__ __forceinline__ const __bf16* pairPtr(const __bf16* cen,
                                                 const __bf16 (*nb)[256], int m, int k) {
  return (k < 256) ? (cen + k) : (&nb[m][0] + (k - 256));
}
__device__ __forceinline__ v16bf loadA_pair(const __bf16* cen,
                                            const __bf16 (*nb)[256], int m, int k) {
  v8bf lo = *(const v8bf*)pairPtr(cen, nb, m, k);
  v8bf hi = *(const v8bf*)pairPtr(cen, nb, m, k + 16);
  return cat8(lo, hi);
}

__device__ __forceinline__ v8f wmma_bf16(v16bf a, v16bf b, v8f c) {
  return __builtin_amdgcn_wmma_f32_16x16x32_bf16(false, a, false, b, (short)0, c,
                                                 false, false);
}

// ---------------------------------------------------------------------------
// Prep: fp32 weights -> bf16 in WMMA-friendly "BT" layouts (row n, contiguous k)
//   w1t  [128][512] : w1t[n][f]  = W1[h][f][a], n = h*64+a
//   gatet[2][128][128] : same layout as gate_W (y = x @ W.T)
//   outWb[256][256]    : same layout as out_W  (y = x @ W.T)
// ---------------------------------------------------------------------------
__global__ __launch_bounds__(256) void prep_weights_kernel(
    const float* __restrict__ W1, const float* __restrict__ gW,
    const float* __restrict__ oW,
    __bf16* __restrict__ w1t, __bf16* __restrict__ gatet,
    __bf16* __restrict__ outWb) {
  int tid = blockIdx.x * 256 + threadIdx.x;
  if (tid < 128 * 512) {
    int n = tid >> 9, f = tid & 511;
    int h = n >> 6, a = n & 63;
    w1t[tid] = (__bf16)W1[((size_t)h * 512 + f) * 64 + a];
  }
  if (tid < 2 * 128 * 128) gatet[tid] = (__bf16)gW[tid];
  if (tid < 256 * 256)     outWb[tid] = (__bf16)oW[tid];
}

// ---------------------------------------------------------------------------
// Main kernel: one block = 16 batch rows, 8 waves.
// Neighbor gathers are double-buffered: global loads for row bb+1 are issued
// before the WMMA work of row bb, hiding random-gather latency under compute.
// ---------------------------------------------------------------------------
__global__ __launch_bounds__(NTHREADS) void nbr_attn_kernel(
    const int*   __restrict__ cidx,   // [B]
    const float* __restrict__ emb,    // [N,256]
    const int*   __restrict__ nidx,   // [N,16]
    const float* __restrict__ nwt,    // [N,16]
    const float* __restrict__ b1,     // [128]  (H*A flattened)
    const float* __restrict__ W2,     // [128]
    const float* __restrict__ b2,     // [2]
    const float* __restrict__ gb,     // [256]  (H*HD flattened)
    const float* __restrict__ ob,     // [256]
    const __bf16* __restrict__ w1t,   // [128][512]
    const __bf16* __restrict__ gatet, // [2][128][128]
    const __bf16* __restrict__ outWb, // [256][256]
    float* __restrict__ out)          // [B,256]
{
  __shared__ float  cenF[TB][256];      // centers fp32
  __shared__ __bf16 cenB[TB][256];      // centers bf16 (gate GEMM A + pair half)
  __shared__ float  aggA[TB][256];      // attention aggregation results
  __shared__ __bf16 combB[TB][256];     // combined (A for out GEMM)
  __shared__ float  nbF[2][16][256];    // neighbor rows fp32 (double buffer)
  __shared__ __bf16 nbB[2][16][256];    // neighbor rows bf16 (double buffer)
  __shared__ float  h1s[16][128];       // GELU(pair@W1+b1) for current b
  __shared__ float  W2s[128];
  __shared__ float  b2s[2];
  __shared__ float  wgt[2][16];         // softmax weights per head
  __shared__ int    cI[TB];

  const int tid  = threadIdx.x;
  const int lane = tid & 31;
  const int wave = tid >> 5;
  const int hf   = lane >> 4;    // lane half (K sub-block select)
  const int l16  = lane & 15;
  const int bbase = blockIdx.x * TB;

  // gather-thread geometry: 16 threads per neighbor row, 4 float4 chunks each
  const int gr = tid >> 4;       // row 0..15
  const int gc = tid & 15;       // chunk lane

  // ---- prologue: center indices, W2/b2 ----
  if (tid < TB) cI[tid] = cidx[bbase + tid];
  if (tid < 128) W2s[tid] = W2[tid];
  if (tid < 2)   b2s[tid] = b2[tid];
  __syncthreads();

  float4 pf0, pf1, pf2, pf3;   // in-flight neighbor gather for next batch row
  {
    // center rows -> LDS (fp32 + bf16)
    int ci = cI[gr];
    bool v = ci >= 0;
    size_t safe = (size_t)(v ? ci : 0);
    const float4* src = (const float4*)(emb + safe * 256);
    for (int j = 0; j < 4; ++j) {
      int cc = gc + j * 16;
      float4 x = v ? src[cc] : make_float4(0.f, 0.f, 0.f, 0.f);
      ((float4*)cenF[gr])[cc] = x;
      cenB[gr][cc * 4 + 0] = (__bf16)x.x;
      cenB[gr][cc * 4 + 1] = (__bf16)x.y;
      cenB[gr][cc * 4 + 2] = (__bf16)x.z;
      cenB[gr][cc * 4 + 3] = (__bf16)x.w;
    }
    // issue neighbor gather for bb = 0 (each thread resolves its own index)
    int c0 = cI[0];
    size_t s0 = (size_t)(c0 < 0 ? 0 : c0);
    int gi = nidx[s0 * 16 + gr];
    const float4* nsrc = (const float4*)(emb + (size_t)(gi < 0 ? 0 : gi) * 256);
    pf0 = nsrc[gc +  0];
    pf1 = nsrc[gc + 16];
    pf2 = nsrc[gc + 32];
    pf3 = nsrc[gc + 48];
  }

  // ---- per-batch-row attention ----
  for (int bb = 0; bb < TB; ++bb) {
    const int cur = bb & 1;

    {  // land the prefetched neighbor rows into buffer `cur` (fp32 + bf16)
      float4 xs[4] = {pf0, pf1, pf2, pf3};
      for (int j = 0; j < 4; ++j) {
        int cc = gc + j * 16;
        float4 x = xs[j];
        ((float4*)nbF[cur][gr])[cc] = x;
        nbB[cur][gr][cc * 4 + 0] = (__bf16)x.x;
        nbB[cur][gr][cc * 4 + 1] = (__bf16)x.y;
        nbB[cur][gr][cc * 4 + 2] = (__bf16)x.z;
        nbB[cur][gr][cc * 4 + 3] = (__bf16)x.w;
      }
    }
    if (bb + 1 < TB) {  // issue gather for bb+1 (completes during the GEMM)
      int cn = cI[bb + 1];
      size_t sn = (size_t)(cn < 0 ? 0 : cn);
      int gi = nidx[sn * 16 + gr];
      const float4* nsrc = (const float4*)(emb + (size_t)(gi < 0 ? 0 : gi) * 256);
      pf0 = nsrc[gc +  0];
      pf1 = nsrc[gc + 16];
      pf2 = nsrc[gc + 32];
      pf3 = nsrc[gc + 48];
    }
    __syncthreads();

    // h1 = GELU(pair[16,512] @ W1T.T + b1) : wave w owns N-tile w (16 cols)
    // A-fragment loads pipelined one k-step ahead of the WMMA chain.
    {
      int n = wave * 16 + l16;
      v8f acc = {};
      const __bf16* Brow = w1t + (size_t)n * 512;
      const __bf16* cen = &cenB[bb][0];
      v16bf a_cur = loadA_pair(cen, nbB[cur], l16, hf * 8);
#pragma unroll
      for (int ks = 0; ks < 16; ++ks) {
        v16bf a_nxt = a_cur;
        if (ks < 15)
          a_nxt = loadA_pair(cen, nbB[cur], l16, (ks + 1) * 32 + hf * 8);
        v16bf b = cat8(*(const v8bf*)(Brow + ks * 32 + hf * 8), *(const v8bf*)(Brow + ks * 32 + hf * 8 + 16));
        acc = wmma_bf16(a_cur, b, acc);
        a_cur = a_nxt;
      }
      float bias = b1[n];
      for (int r = 0; r < 8; ++r) {
        int m = r + hf * 8;
        float x = acc[r] + bias;
        h1s[m][n] = 0.5f * x * (1.0f + erff(x * 0.70710678118654752f));  // exact GELU
      }
    }
    __syncthreads();

    // logits + masked softmax over K=16 (wave 0: lane = h*16 + k)
    if (wave == 0) {
      int k = l16, h = hf;
      float dot = 0.f;
      for (int a = 0; a < 64; ++a) dot += h1s[k][h * 64 + a] * W2s[h * 64 + a];
      int ci = cI[bb];
      size_t s = (size_t)(ci < 0 ? 0 : ci);
      int gi = nidx[s * 16 + k];
      float wv = nwt[s * 16 + k];
      bool v = gi >= 0;
      float lg = dot + b2s[h] + __logf(wv + 1e-8f);
      if (!v) lg = -3.402823466e38f;
      int vc = v ? 1 : 0;
      float mx = lg;
      for (int m = 1; m < 16; m <<= 1) {
        mx = fmaxf(mx, __shfl_xor(mx, m, 32));
        vc += __shfl_xor(vc, m, 32);
      }
      float e = (vc > 0) ? __expf(lg - mx) : 1.0f;  // no valid nbrs -> uniform
      float sm = e;
      for (int m = 1; m < 16; m <<= 1) sm += __shfl_xor(sm, m, 32);
      wgt[h][k] = e / sm;
    }
    __syncthreads();

    // agg[h][d] = sum_k w[h][k] * nbF[k][h*128+d]  (fp32, 1 output/thread)
    {
      int h = tid >> 7;
      float acc = 0.f;
      for (int k = 0; k < 16; ++k) acc += wgt[h][k] * nbF[cur][k][tid];
      aggA[bb][tid] = acc;
    }
    // no trailing barrier: next iteration writes the OTHER nb buffer and
    // synchronizes before any shared reads.
  }

  __syncthreads();

  // ---- gate GEMM (2 heads x [16,128]x[128,128]) + combine ----
  for (int i = 0; i < 2; ++i) {
    int tg = wave * 2 + i;          // 0..15
    int h = tg >> 3, nt = tg & 7;
    int n = nt * 16 + l16;          // e in [0,128)
    v8f acc = {};
    const __bf16* Brow = gatet + ((size_t)h * 128 + n) * 128;
    v16bf a_cur = loadA_contig(&cenB[0][0], 256, l16, h * 128 + hf * 8);
    v16bf b_cur = cat8(*(const v8bf*)(Brow + hf * 8), *(const v8bf*)(Brow + hf * 8 + 16));
#pragma unroll
    for (int ks = 0; ks < 4; ++ks) {
      v16bf a_nxt = a_cur, b_nxt = b_cur;
      if (ks < 3) {
        a_nxt = loadA_contig(&cenB[0][0], 256, l16, h * 128 + (ks + 1) * 32 + hf * 8);
        b_nxt = cat8(*(const v8bf*)(Brow + (ks + 1) * 32 + hf * 8), *(const v8bf*)(Brow + (ks + 1) * 32 + hf * 8 + 16));
      }
      acc = wmma_bf16(a_cur, b_cur, acc);
      a_cur = a_nxt; b_cur = b_nxt;
    }
    int col = h * 128 + n;
    float gbv = gb[col];
    for (int r = 0; r < 8; ++r) {
      int m = r + hf * 8;
      float g = 1.0f / (1.0f + __expf(-(acc[r] + gbv)));
      float comb = cenF[m][col] + g * aggA[m][col];
      combB[m][col] = (__bf16)comb;
    }
  }
  __syncthreads();

  // ---- out projection: [16,256] x [256,256] ----
  for (int i = 0; i < 2; ++i) {
    int nt = wave * 2 + i;          // 0..15
    int n = nt * 16 + l16;          // j in [0,256)
    v8f acc = {};
    const __bf16* Brow = outWb + (size_t)n * 256;
    v16bf a_cur = loadA_contig(&combB[0][0], 256, l16, hf * 8);
    v16bf b_cur = cat8(*(const v8bf*)(Brow + hf * 8), *(const v8bf*)(Brow + hf * 8 + 16));
#pragma unroll
    for (int ks = 0; ks < 8; ++ks) {
      v16bf a_nxt = a_cur, b_nxt = b_cur;
      if (ks < 7) {
        a_nxt = loadA_contig(&combB[0][0], 256, l16, (ks + 1) * 32 + hf * 8);
        b_nxt = cat8(*(const v8bf*)(Brow + (ks + 1) * 32 + hf * 8), *(const v8bf*)(Brow + (ks + 1) * 32 + hf * 8 + 16));
      }
      acc = wmma_bf16(a_cur, b_cur, acc);
      a_cur = a_nxt; b_cur = b_nxt;
    }
    float obv = ob[n];
    for (int r = 0; r < 8; ++r) {
      int m = r + hf * 8;
      bool v = cI[m] >= 0;
      out[(size_t)(bbase + m) * 256 + n] = v ? (acc[r] + obv) : 0.0f;
    }
  }
}

// ---------------------------------------------------------------------------
extern "C" void kernel_launch(void* const* d_in, const int* in_sizes, int n_in,
                              void* d_out, int out_size, void* d_ws, size_t ws_size,
                              hipStream_t stream) {
  const int*   cidx = (const int*)d_in[0];
  const float* emb  = (const float*)d_in[1];
  const int*   nidx = (const int*)d_in[2];
  const float* nwt  = (const float*)d_in[3];
  const float* W1   = (const float*)d_in[4];
  const float* b1   = (const float*)d_in[5];
  const float* W2   = (const float*)d_in[6];
  const float* b2   = (const float*)d_in[7];
  const float* gW   = (const float*)d_in[8];
  const float* gb   = (const float*)d_in[9];
  const float* oW   = (const float*)d_in[10];
  const float* ob   = (const float*)d_in[11];

  char* ws = (char*)d_ws;
  __bf16* w1t   = (__bf16*)(ws);            // 128*512*2  = 131072 B
  __bf16* gatet = (__bf16*)(ws + 131072);   // 2*128*128*2 = 65536 B
  __bf16* outWb = (__bf16*)(ws + 196608);   // 256*256*2  = 131072 B

  prep_weights_kernel<<<256, 256, 0, stream>>>(W1, gW, oW, w1t, gatet, outWb);

  const int Bsz = in_sizes[0];              // 16384
  nbr_attn_kernel<<<Bsz / TB, NTHREADS, 0, stream>>>(
      cidx, emb, nidx, nwt, b1, W2, b2, gb, ob, w1t, gatet, outWb,
      (float*)d_out);
}
